// MultiHeadAttentionWrapper_89816356094700
// MI455X (gfx1250) — matrix operand order repaired and checked by reading, not compiled, not run on hardware
//
#include <hip/hip_runtime.h>
#include <hip/hip_bf16.h>
#include <stdint.h>

// Problem constants (from reference): B=2, S=2048, D=1024, H=16, DQ=DV=64
#define B_  2
#define S_  2048
#define D_  1024
#define H_  16
#define DH_ 64

typedef _Float16 half_t;
typedef __attribute__((ext_vector_type(16))) _Float16 v16h;
typedef __attribute__((ext_vector_type(8)))  _Float16 v8h;
typedef __attribute__((ext_vector_type(8)))  float    v8f;

// ---------------------------------------------------------------------------
// WMMA helper (CDNA5 wave32, 16x16x32 f16 -> f32)
// ---------------------------------------------------------------------------
__device__ __forceinline__ v8f wmma_f16(v16h a, v16h b, v8f c) {
  return __builtin_amdgcn_wmma_f32_16x16x32_f16(false, a, false, b, (short)0, c,
                                                false, false);
}

// A fragment: 16x32 (M x K), source row-major with leading dim `ld` (halves).
// Lane r (=lane&15) holds row r; half-wave hi selects K phase:
//   elements 0..15 -> K = 16*hi + {0..15}
__device__ __forceinline__ v16h load_a_frag(const half_t* __restrict__ base, int ld) {
  const int lane = threadIdx.x & 31;
  const int r = lane & 15, hi = lane >> 4;
  const half_t* p = base + (size_t)r * ld + 16 * hi;
  v8h lo = *(const v8h*)p;
  v8h hh = *(const v8h*)(p + 8);
  v16h out;
#pragma unroll
  for (int i = 0; i < 8; ++i) { out[i] = lo[i]; out[i + 8] = hh[i]; }
  return out;
}

// B fragment: 32x16 (K x N) from row-major Bsrc[n][k] (B^T in memory).
// Lane r holds column n=r; half-wave hi selects K = 16*hi + {0..15}.
__device__ __forceinline__ v16h load_b_frag(const half_t* __restrict__ base, int ld) {
  const int lane = threadIdx.x & 31;
  const int r = lane & 15, hi = lane >> 4;
  return *(const v16h*)(base + (size_t)r * ld + 16 * hi);
}

// ---------------------------------------------------------------------------
// f32 -> f16 conversion (grid-stride)
// ---------------------------------------------------------------------------
__global__ void cvt_f32_to_f16(const float* __restrict__ src,
                               half_t* __restrict__ dst, size_t n) {
  size_t i = (size_t)blockIdx.x * blockDim.x + threadIdx.x;
  size_t stride = (size_t)gridDim.x * blockDim.x;
  for (; i < n; i += stride) dst[i] = (half_t)src[i];
}

// ---------------------------------------------------------------------------
// QKV projection: per wave, one (b,h, 32-row block of S).
// Two x A-fragments feed 24 WMMAs per k-step (Q,K,V x 4 n-tiles x 2 m-tiles);
// every weight B-fragment is reused across both m-tiles (halves L2 traffic).
// Q,K stored row-major [S,64]; V stored transposed Vt[64][S].
// ---------------------------------------------------------------------------
__global__ __launch_bounds__(128) void qkv_kernel(
    const half_t* __restrict__ xh, const half_t* __restrict__ wqh,
    const half_t* __restrict__ wkh, const half_t* __restrict__ wvh,
    half_t* __restrict__ Qb, half_t* __restrict__ Kb, half_t* __restrict__ Vtb) {
  const int w = blockIdx.x * (blockDim.x >> 5) + (threadIdx.x >> 5);
  const int mblk = w & (S_ / 32 - 1);
  const int bh = w / (S_ / 32);           // b*H + h
  const int h = bh % H_;
  const int m0 = mblk * 32;
  const int lane = threadIdx.x & 31, r = lane & 15, hi = lane >> 4;

  const half_t* xa = xh + ((size_t)(bh / H_) * S_ + m0) * D_;
  const half_t* wq = wqh + (size_t)h * DH_ * D_;
  const half_t* wk = wkh + (size_t)h * DH_ * D_;
  const half_t* wv = wvh + (size_t)h * DH_ * D_;

  v8f aq[2][4] = {}, ak[2][4] = {}, av[2][4] = {};

  for (int kk = 0; kk < D_; kk += 32) {
    v16h a0 = load_a_frag(xa + kk, D_);
    v16h a1 = load_a_frag(xa + (size_t)16 * D_ + kk, D_);
    if (kk + 32 < D_)  // prefetch next x A-tile (rows m0..m0+31, col kk+32)
      __builtin_prefetch(xa + (size_t)lane * D_ + kk + 32, 0, 0);
#pragma unroll
    for (int t = 0; t < 4; ++t) {
      v16h bqf = load_b_frag(wq + (size_t)(16 * t) * D_ + kk, D_);
      v16h bkf = load_b_frag(wk + (size_t)(16 * t) * D_ + kk, D_);
      v16h bvf = load_b_frag(wv + (size_t)(16 * t) * D_ + kk, D_);
      aq[0][t] = wmma_f16(a0, bqf, aq[0][t]);
      aq[1][t] = wmma_f16(a1, bqf, aq[1][t]);
      ak[0][t] = wmma_f16(a0, bkf, ak[0][t]);
      ak[1][t] = wmma_f16(a1, bkf, ak[1][t]);
      av[0][t] = wmma_f16(a0, bvf, av[0][t]);
      av[1][t] = wmma_f16(a1, bvf, av[1][t]);
    }
  }

  half_t* Qo = Qb + (size_t)bh * S_ * DH_;
  half_t* Ko = Kb + (size_t)bh * S_ * DH_;
#pragma unroll
  for (int mi = 0; mi < 2; ++mi)
#pragma unroll
    for (int t = 0; t < 4; ++t)
#pragma unroll
      for (int j = 0; j < 8; ++j) {
        const int s = m0 + 16 * mi + 8 * hi + j;
        const int dq = 16 * t + r;
        Qo[(size_t)s * DH_ + dq] = (half_t)aq[mi][t][j];
        Ko[(size_t)s * DH_ + dq] = (half_t)ak[mi][t][j];
      }

  half_t* Vo = Vtb + (size_t)bh * DH_ * S_;
#pragma unroll
  for (int mi = 0; mi < 2; ++mi)
#pragma unroll
    for (int t = 0; t < 4; ++t) {
      v8h pk;
#pragma unroll
      for (int j = 0; j < 8; ++j) pk[j] = (half_t)av[mi][t][j];
      *(v8h*)(Vo + (size_t)(16 * t + r) * S_ + m0 + 16 * mi + 8 * hi) = pk;
    }
}

// ---------------------------------------------------------------------------
// P^T fragment builder: p = exp(s - mn) in f16; own lane has rows 8*hi+{0..7}
// of each 16-key tile, partner (xor 16) supplies the other 8 rows.
// ---------------------------------------------------------------------------
__device__ __forceinline__ v16h softmax_frag(v8f s0, v8f s1, float mn,
                                             float& psum, int hi) {
  union { v8h h; uint32_t u[4]; } P0, P1, P0p, P1p;
  float ps = 0.f;
#pragma unroll
  for (int j = 0; j < 8; ++j) {
    const float p0 = __expf(s0[j] - mn);
    const float p1 = __expf(s1[j] - mn);
    ps += p0 + p1;
    P0.h[j] = (half_t)p0;
    P1.h[j] = (half_t)p1;
  }
  psum = ps;
#pragma unroll
  for (int i = 0; i < 4; ++i) {
    P0p.u[i] = __shfl_xor(P0.u[i], 16);
    P1p.u[i] = __shfl_xor(P1.u[i], 16);
  }
  v16h bp;
#pragma unroll
  for (int j = 0; j < 8; ++j) {
    bp[j]     = hi ? P1p.h[j] : P0.h[j];   // K = 16*hi + {0..7}
    bp[j + 8] = hi ? P1.h[j]  : P0p.h[j];  // K = 16*hi + {8..15}
  }
  return bp;
}

// ---------------------------------------------------------------------------
// Transposed flash attention: per wave, one (b,h, 32-query block).
//   S^T = K * Q^T  (D-tile column = query -> softmax stats live in one lane
//   plus its xor-16 partner);  out^T = V^T * P^T accumulated in f32.
// K/V fragments are shared across the two 16-query sub-blocks.
// ---------------------------------------------------------------------------
__global__ __launch_bounds__(128) void attn_kernel(
    const half_t* __restrict__ Qb, const half_t* __restrict__ Kb,
    const half_t* __restrict__ Vtb, half_t* __restrict__ catb) {
  const int w = blockIdx.x * (blockDim.x >> 5) + (threadIdx.x >> 5);
  const int mblk = w & (S_ / 32 - 1);
  const int bh = w / (S_ / 32);
  const int m0 = mblk * 32;
  const int lane = threadIdx.x & 31, r = lane & 15, hi = lane >> 4;

  const half_t* Qp = Qb + (size_t)bh * S_ * DH_;
  const half_t* Kp = Kb + (size_t)bh * S_ * DH_;
  const half_t* Vp = Vtb + (size_t)bh * DH_ * S_;
  half_t* cp = catb + ((size_t)(bh / H_) * S_) * (H_ * DH_) + (bh % H_) * DH_;

  // Q^T B-fragments for the two 16-query sub-blocks, loaded once.
  v16h bq0[2], bq1[2];
#pragma unroll
  for (int i = 0; i < 2; ++i) {
    bq0[i] = load_b_frag(Qp + (size_t)(m0 + 16 * i) * DH_ + 0, DH_);
    bq1[i] = load_b_frag(Qp + (size_t)(m0 + 16 * i) * DH_ + 32, DH_);
  }

  const float NEG_INF = -__builtin_inff();
  float mrun[2] = { NEG_INF, NEG_INF };
  float lrun[2] = { 0.f, 0.f };
  v8f acc[2][4] = {};
  const int mq[2] = { m0 + r, m0 + 16 + r };  // lane's query index per block
  const float scale = 0.125f;                 // 1/sqrt(64)

  for (int tb = 0; tb < m0 + 32; tb += 32) {
    // Shared K fragments for two 16-key tiles
    v16h ka0 = load_a_frag(Kp + (size_t)tb * DH_, DH_);
    v16h ka1 = load_a_frag(Kp + (size_t)tb * DH_ + 32, DH_);
    v16h kb0 = load_a_frag(Kp + (size_t)(tb + 16) * DH_, DH_);
    v16h kb1 = load_a_frag(Kp + (size_t)(tb + 16) * DH_ + 32, DH_);
    // Shared V^T fragments (rows dv, K = t local)
    v16h av0 = load_a_frag(Vp + (size_t)0  * S_ + tb, S_);
    v16h av1 = load_a_frag(Vp + (size_t)16 * S_ + tb, S_);
    v16h av2 = load_a_frag(Vp + (size_t)32 * S_ + tb, S_);
    v16h av3 = load_a_frag(Vp + (size_t)48 * S_ + tb, S_);

    if (tb + 32 < m0 + 32) {   // prefetch next K/V tiles (gfx1250 prefetch)
      __builtin_prefetch(Kp + (size_t)(tb + 32 + lane) * DH_, 0, 0);
      __builtin_prefetch(Vp + (size_t)lane * S_ + tb + 32, 0, 0);
      __builtin_prefetch(Vp + (size_t)(32 + lane) * S_ + tb + 32, 0, 0);
    }

#pragma unroll
    for (int i = 0; i < 2; ++i) {
      v8f s0 = {}, s1 = {};
      s0 = wmma_f16(ka0, bq0[i], s0);
      s0 = wmma_f16(ka1, bq1[i], s0);
      s1 = wmma_f16(kb0, bq0[i], s1);
      s1 = wmma_f16(kb1, bq1[i], s1);

      // scale + causal mask (key t <= query m); per-lane tile max
      float tmax = NEG_INF;
#pragma unroll
      for (int j = 0; j < 8; ++j) {
        const int t0 = tb + 8 * hi + j;
        const int t1 = t0 + 16;
        const float x0 = (t0 <= mq[i]) ? s0[j] * scale : NEG_INF;
        const float x1 = (t1 <= mq[i]) ? s1[j] * scale : NEG_INF;
        s0[j] = x0; s1[j] = x1;
        tmax = fmaxf(tmax, fmaxf(x0, x1));
      }
      tmax = fmaxf(tmax, __shfl_xor(tmax, 16));
      const float mn = fmaxf(mrun[i], tmax);
      const float alpha = __expf(mrun[i] - mn);

      float ps;
      v16h bp = softmax_frag(s0, s1, mn, ps, hi);
      lrun[i] = lrun[i] * alpha + ps + __shfl_xor(ps, 16);
      mrun[i] = mn;

#pragma unroll
      for (int a = 0; a < 4; ++a)
#pragma unroll
        for (int j = 0; j < 8; ++j) acc[i][a][j] *= alpha;

      acc[i][0] = wmma_f16(av0, bp, acc[i][0]);
      acc[i][1] = wmma_f16(av1, bp, acc[i][1]);
      acc[i][2] = wmma_f16(av2, bp, acc[i][2]);
      acc[i][3] = wmma_f16(av3, bp, acc[i][3]);
    }
  }

  // Epilogue: divide by softmax denom, store cat[s][h*64+dv] (16B stores)
#pragma unroll
  for (int i = 0; i < 2; ++i) {
    const float invl = 1.0f / lrun[i];
    const size_t srow = (size_t)(m0 + 16 * i + r) * (H_ * DH_);
#pragma unroll
    for (int a = 0; a < 4; ++a) {
      v8h o;
#pragma unroll
      for (int j = 0; j < 8; ++j) o[j] = (half_t)(acc[i][a][j] * invl);
      *(v8h*)(cp + srow + 16 * a + 8 * hi) = o;
    }
  }
}

// ---------------------------------------------------------------------------
// Output projection: out[bs][d] = sum_f cat[bs][f] * w_o[d][f]  (f32 out)
// Per wave: 32 rows x 64 cols; w_o B-fragments reused across both row tiles.
// ---------------------------------------------------------------------------
__global__ __launch_bounds__(128) void oproj_kernel(
    const half_t* __restrict__ catb, const half_t* __restrict__ woh,
    float* __restrict__ out) {
  const int w = blockIdx.x * (blockDim.x >> 5) + (threadIdx.x >> 5);
  const int dblk = w & 15;
  const int mblk = w >> 4;
  const int m0 = mblk * 32, d0 = dblk * 64;
  const int lane = threadIdx.x & 31, r = lane & 15, hi = lane >> 4;

  const half_t* arow = catb + (size_t)m0 * (H_ * DH_);
  v8f acc[2][4] = {};

  for (int kk = 0; kk < H_ * DH_; kk += 32) {
    v16h a0 = load_a_frag(arow + kk, H_ * DH_);
    v16h a1 = load_a_frag(arow + (size_t)16 * (H_ * DH_) + kk, H_ * DH_);
    if (kk + 32 < H_ * DH_)
      __builtin_prefetch(arow + (size_t)lane * (H_ * DH_) + kk + 32, 0, 0);
#pragma unroll
    for (int t = 0; t < 4; ++t) {
      v16h bw = load_b_frag(woh + (size_t)(d0 + 16 * t) * D_ + kk, D_);
      acc[0][t] = wmma_f16(a0, bw, acc[0][t]);
      acc[1][t] = wmma_f16(a1, bw, acc[1][t]);
    }
  }

#pragma unroll
  for (int mi = 0; mi < 2; ++mi)
#pragma unroll
    for (int t = 0; t < 4; ++t)
#pragma unroll
      for (int j = 0; j < 8; ++j) {
        const int s = m0 + 16 * mi + 8 * hi + j;
        const int d = d0 + 16 * t + r;
        out[(size_t)s * D_ + d] = acc[mi][t][j];
      }
}

// ---------------------------------------------------------------------------
// Host launcher
// ---------------------------------------------------------------------------
extern "C" void kernel_launch(void* const* d_in, const int* in_sizes, int n_in,
                              void* d_out, int out_size, void* d_ws, size_t ws_size,
                              hipStream_t stream) {
  (void)in_sizes; (void)n_in; (void)out_size; (void)ws_size;
  const float* x  = (const float*)d_in[0];   // [B,S,D]
  const float* wq = (const float*)d_in[1];   // [H,64,D]
  const float* wk = (const float*)d_in[2];   // [H,64,D]
  const float* wv = (const float*)d_in[3];   // [H,64,D]
  const float* wo = (const float*)d_in[4];   // [D, H*64]
  float* out = (float*)d_out;                // [B,S,D]

  // Workspace layout (f16 halves); total = 48 MiB
  const size_t N_X  = (size_t)B_ * S_ * D_;        // 4,194,304
  const size_t N_W  = (size_t)H_ * DH_ * D_;       // 1,048,576
  const size_t N_WO = (size_t)D_ * H_ * DH_;       // 1,048,576
  const size_t N_QK = (size_t)B_ * H_ * S_ * DH_;  // 4,194,304

  half_t* xh   = (half_t*)d_ws;
  half_t* wqh  = xh + N_X;
  half_t* wkh  = wqh + N_W;
  half_t* wvh  = wkh + N_W;
  half_t* woh  = wvh + N_W;
  half_t* Qb   = woh + N_WO;
  half_t* Kb   = Qb + N_QK;
  half_t* Vtb  = Kb + N_QK;
  half_t* catb = Vtb + N_QK;

  cvt_f32_to_f16<<<1024, 256, 0, stream>>>(x,  xh,  N_X);
  cvt_f32_to_f16<<<256,  256, 0, stream>>>(wq, wqh, N_W);
  cvt_f32_to_f16<<<256,  256, 0, stream>>>(wk, wkh, N_W);
  cvt_f32_to_f16<<<256,  256, 0, stream>>>(wv, wvh, N_W);
  cvt_f32_to_f16<<<256,  256, 0, stream>>>(wo, woh, N_WO);

  // B*H*(S/32) = 2048 waves, 4 waves (128 threads) per block
  qkv_kernel<<<512, 128, 0, stream>>>(xh, wqh, wkh, wvh, Qb, Kb, Vtb);
  attn_kernel<<<512, 128, 0, stream>>>(Qb, Kb, Vtb, catb);
  // (B*S/32)*(D/64) = 2048 waves
  oproj_kernel<<<512, 128, 0, stream>>>(catb, woh, out);
}
